// QAttn_87729001988399
// MI455X (gfx1250) — compile-verified, matrix operand order checked by reading, not run on hardware
//
#include <hip/hip_runtime.h>

// ---------------------------------------------------------------------------
// HEPT attention block for MI455X (gfx1250, wave32, WMMA 16x16x32 f16).
// N=3072 points, D=64 per head, H=8 heads, HD=512, C=3 coords, K=8.
// ---------------------------------------------------------------------------

typedef __attribute__((ext_vector_type(16))) _Float16 v16h;
typedef __attribute__((ext_vector_type(8)))  float    v8f;

#define NP   3072
#define DIM  64
#define HEADS 8
#define HDIM 512
#define CORD 3

// ---------------------------------------------------------------------------
// WMMA wrapper: D = A(16x32 f16) * B(32x16 f16) + C(16x16 f32)
// ---------------------------------------------------------------------------
__device__ __forceinline__ v8f wmma_f16(v16h a, v16h b, v8f c) {
  return __builtin_amdgcn_wmma_f32_16x16x32_f16(
      /*neg_a=*/false, a, /*neg_b=*/false, b,
      /*c_mod=*/(short)0, c, /*reuse_a=*/false, /*reuse_b=*/false);
}

// A-fragment (16x32, f16) from a row-major matrix: rows = M, cols = K.
// Lane l<16: row M=l, K = {0..7, 16..23}; lane l>=16: row M=l-16, K={8..15,24..31}.
// Also serves as a B-fragment when B[k][n] = Mat[n][k] (i.e. Mat stored [N x K]).
__device__ __forceinline__ v16h frag_rowmajor(const _Float16* p, int ld,
                                              int row_base, int col_base) {
  const int l  = threadIdx.x & 31;
  const int m  = l & 15;
  const int of = (l >> 4) * 8;
  const _Float16* rp = p + (size_t)(row_base + m) * ld + col_base + of;
  v16h a;
#pragma unroll
  for (int e = 0; e < 8; ++e) { a[e] = rp[e]; a[e + 8] = rp[e + 16]; }
  return a;
}

// C/D accumulator element (r, lane): row M = r + 8*(lane>>4), col N = lane&15.
__device__ __forceinline__ void store_acc_f16(_Float16* p, int ld, int row_base,
                                              int col_base, v8f c) {
  const int l  = threadIdx.x & 31;
  const int n  = l & 15;
  const int mo = (l >> 4) * 8;
#pragma unroll
  for (int r = 0; r < 8; ++r)
    p[(size_t)(row_base + mo + r) * ld + col_base + n] = (_Float16)c[r];
}

// Same, but store TRANSPOSED into a [cols x rows] matrix (ldT = #rows).
// Per (lane,nc) the 8 r-elements land contiguously -> 16B vectorizable stores.
__device__ __forceinline__ void store_acc_f16_T(_Float16* p, int ldT, int row_base,
                                                int col_base, v8f c) {
  const int l  = threadIdx.x & 31;
  const int n  = l & 15;
  const int mo = (l >> 4) * 8;
  _Float16* cp = p + (size_t)(col_base + n) * ldT + row_base + mo;
#pragma unroll
  for (int r = 0; r < 8; ++r) cp[r] = (_Float16)c[r];
}

// half-wave (16-lane) row reductions for the C/D layout
__device__ __forceinline__ float hmax16(float x) {
#pragma unroll
  for (int m = 8; m > 0; m >>= 1) x = fmaxf(x, __shfl_xor(x, m, 16));
  return x;
}
__device__ __forceinline__ float hsum16(float x) {
#pragma unroll
  for (int m = 8; m > 0; m >>= 1) x += __shfl_xor(x, m, 16);
  return x;
}
__device__ __forceinline__ float wsum32(float x) {
#pragma unroll
  for (int m = 16; m > 0; m >>= 1) x += __shfl_xor(x, m, 32);
  return x;
}

// ---------------------------------------------------------------------------
// Kernel 0a: int "codes" -> f16 dequant:  (w - 8) * scale * extra
// ---------------------------------------------------------------------------
__global__ void k_dequant(const int* __restrict__ w, const float* __restrict__ sc,
                          float extra, _Float16* __restrict__ out, int n) {
  int i = blockIdx.x * blockDim.x + threadIdx.x;
  if (i < n) out[i] = (_Float16)(((float)w[i] - 8.0f) * sc[0] * extra);
}

// Kernel 0b: f32 -> f16 convert (for wo)
__global__ void k_f32_to_f16(const float* __restrict__ in, _Float16* __restrict__ out,
                             int n) {
  int i = blockIdx.x * blockDim.x + threadIdx.x;
  if (i < n) out[i] = (_Float16)in[i];
}

// Kernel 0c: RPE distance weights  w[h][r] = mean_{d,k} |deq(w_rpe[h*64+d][r*8+k])|
__global__ void k_rpe(const int* __restrict__ wr, const float* __restrict__ sc,
                      float* __restrict__ out) {
  int t = threadIdx.x;
  if (t < HEADS * 2) {
    int h = t >> 1, r = t & 1;
    float s = sc[0], acc = 0.0f;
    for (int d = 0; d < DIM; ++d)
      for (int k = 0; k < 8; ++k)
        acc += fabsf(((float)wr[(h * DIM + d) * 16 + r * 8 + k] - 8.0f) * s);
    out[t] = acc * (1.0f / 512.0f);
  }
}

// ---------------------------------------------------------------------------
// Kernel 1: LayerNorm over D=64, output f16.  One wave per row (8 rows/block).
// ---------------------------------------------------------------------------
__global__ void k_layernorm(const float* __restrict__ x, const float* __restrict__ g,
                            const float* __restrict__ b, _Float16* __restrict__ out) {
  const int lane = threadIdx.x & 31;
  const int row  = blockIdx.x * 8 + (threadIdx.x >> 5);
  const float* xr = x + (size_t)row * DIM;
  float a0 = xr[lane], a1 = xr[lane + 32];
  float mean = wsum32(a0 + a1) * (1.0f / 64.0f);
  float d0 = a0 - mean, d1 = a1 - mean;
  float var = wsum32(d0 * d0 + d1 * d1) * (1.0f / 64.0f);
  float rs = rsqrtf(var + 1e-5f);
  _Float16* o = out + (size_t)row * DIM;
  o[lane]      = (_Float16)(d0 * rs * g[lane]      + b[lane]);
  o[lane + 32] = (_Float16)(d1 * rs * g[lane + 32] + b[lane + 32]);
}

// ---------------------------------------------------------------------------
// Kernel 2: Q/K/V projection GEMM.  out[N][512] = xn[N][64] @ W[512][64]^T.
// grid = (N/16, HD/64, 3); one wave computes a 16x64 tile (8 WMMAs).
// z==2 (V) stores TRANSPOSED into vt[HD][N] so the attention P*V B-fragments
// become contiguous (k runs along memory) instead of strided gathers.
// ---------------------------------------------------------------------------
__global__ void k_qkv_gemm(const _Float16* __restrict__ xn,
                           const _Float16* __restrict__ wq,
                           const _Float16* __restrict__ wk,
                           const _Float16* __restrict__ wv,
                           _Float16* __restrict__ qo, _Float16* __restrict__ ko,
                           _Float16* __restrict__ vt) {
  const int row0 = blockIdx.x * 16;
  const int col0 = blockIdx.y * 64;
  const _Float16* W = (blockIdx.z == 0) ? wq : (blockIdx.z == 1) ? wk : wv;

  v8f acc[4] = {{}, {}, {}, {}};
#pragma unroll
  for (int kc = 0; kc < 2; ++kc) {
    v16h a = frag_rowmajor(xn, DIM, row0, kc * 32);
#pragma unroll
    for (int nc = 0; nc < 4; ++nc) {
      // B[k][n] = W[col0+nc*16+n][kc*32+k]
      v16h b = frag_rowmajor(W, DIM, col0 + nc * 16, kc * 32);
      acc[nc] = wmma_f16(a, b, acc[nc]);
    }
  }
  if (blockIdx.z == 2) {
#pragma unroll
    for (int nc = 0; nc < 4; ++nc)
      store_acc_f16_T(vt, NP, row0, col0 + nc * 16, acc[nc]);
  } else {
    _Float16* O = (blockIdx.z == 0) ? qo : ko;
#pragma unroll
    for (int nc = 0; nc < 4; ++nc)
      store_acc_f16(O, HDIM, row0, col0 + nc * 16, acc[nc]);
  }
}

// ---------------------------------------------------------------------------
// Kernel 3: flash attention, grid = (N/16, H), one wave per block.
// Per 32-key chunk: 4 WMMAs for S, online softmax, P via LDS, 4 WMMAs for P*V.
// K is read row-major ([key][dim]); V is read from vt[HD][N] (k contiguous).
// Next chunk's K/Vt lines are prefetched (global_prefetch_b8) to overlap with
// the softmax VALU that co-executes with the XDL WMMAs.
// ---------------------------------------------------------------------------
__global__ void k_attention(const _Float16* __restrict__ q,
                            const _Float16* __restrict__ kk,
                            const _Float16* __restrict__ vt,
                            const float* __restrict__ coords,
                            const float* __restrict__ wrpe,
                            _Float16* __restrict__ aggr) {
  __shared__ _Float16 lds[16 * 32];
  const int lane = threadIdx.x & 31;
  const int n    = lane & 15;
  const int mo   = (lane >> 4) * 8;
  const int q0   = blockIdx.x * 16;
  const int h    = blockIdx.y;
  const int hc   = h * DIM;

  // Q tile: 16 rows x 64 dims -> two A fragments (1/sqrt(D) folded into Wq).
  v16h qa0 = frag_rowmajor(q, HDIM, q0, hc);
  v16h qa1 = frag_rowmajor(q, HDIM, q0, hc + 32);

  // per-lane query coords for its 8 rows of the C-layout
  float cq0[8], cq1[8];
#pragma unroll
  for (int r = 0; r < 8; ++r) {
    cq0[r] = coords[(size_t)(q0 + mo + r) * CORD + 0];
    cq1[r] = coords[(size_t)(q0 + mo + r) * CORD + 1];
  }
  const float w0 = wrpe[h * 2 + 0], w1 = wrpe[h * 2 + 1];

  float mrun[8], lrun[8];
#pragma unroll
  for (int r = 0; r < 8; ++r) { mrun[r] = -3.0e38f; lrun[r] = 0.0f; }
  v8f o[4] = {{}, {}, {}, {}};

  for (int kv = 0; kv < NP; kv += 32) {
    // prefetch next chunk: K rows [kv+32, kv+64) and Vt columns kv+32..
    if (kv + 32 < NP) {
      __builtin_prefetch(kk + (size_t)(kv + 32 + lane) * HDIM + hc, 0, 3);
      __builtin_prefetch(vt + (size_t)(hc + (lane & 63)) * NP + kv + 32, 0, 3);
    }

    // S tiles for keys [kv, kv+16) and [kv+16, kv+32): B[k][n] = K[key][d]
    v8f s0 = {}, s1 = {};
    s0 = wmma_f16(qa0, frag_rowmajor(kk, HDIM, kv,      hc),      s0);
    s0 = wmma_f16(qa1, frag_rowmajor(kk, HDIM, kv,      hc + 32), s0);
    s1 = wmma_f16(qa0, frag_rowmajor(kk, HDIM, kv + 16, hc),      s1);
    s1 = wmma_f16(qa1, frag_rowmajor(kk, HDIM, kv + 16, hc + 32), s1);

    // RPE bias: key coords for this lane's column
    float ck0a = coords[(size_t)(kv + n) * CORD + 0];
    float ck1a = coords[(size_t)(kv + n) * CORD + 1];
    float ck0b = coords[(size_t)(kv + 16 + n) * CORD + 0];
    float ck1b = coords[(size_t)(kv + 16 + n) * CORD + 1];

    float cmax[8];
#pragma unroll
    for (int r = 0; r < 8; ++r) {
      float d0 = cq0[r] - ck0a, d1 = cq1[r] - ck1a;
      s0[r] -= w0 * d0 * d0 + w1 * d1 * d1;
      d0 = cq0[r] - ck0b; d1 = cq1[r] - ck1b;
      s1[r] -= w0 * d0 * d0 + w1 * d1 * d1;
      cmax[r] = hmax16(fmaxf(s0[r], s1[r]));
    }

    // online softmax update + stage P (f16) to LDS in 16x32 row-major
#pragma unroll
    for (int r = 0; r < 8; ++r) {
      float mnew  = fmaxf(mrun[r], cmax[r]);
      float alpha = __expf(mrun[r] - mnew);
      float p0 = __expf(s0[r] - mnew);
      float p1 = __expf(s1[r] - mnew);
      lrun[r] = lrun[r] * alpha + hsum16(p0 + p1);
      mrun[r] = mnew;
#pragma unroll
      for (int nc = 0; nc < 4; ++nc) o[nc][r] *= alpha;
      lds[(mo + r) * 32 + n]      = (_Float16)p0;
      lds[(mo + r) * 32 + 16 + n] = (_Float16)p1;
    }
    __syncthreads();

    // P (16x32) as A-fragment; V B-frag from vt: B[k][nn] = vt[hc+d0+nn][kv+k]
    v16h pa = frag_rowmajor(lds, 32, 0, 0);
#pragma unroll
    for (int nc = 0; nc < 4; ++nc) {
      v16h bv = frag_rowmajor(vt, NP, hc + nc * 16, kv);
      o[nc] = wmma_f16(pa, bv, o[nc]);
    }
    __syncthreads();
  }

  // normalize and store aggregated head output (f16) into aggr[N][512]
#pragma unroll
  for (int r = 0; r < 8; ++r) {
    float inv = 1.0f / lrun[r];
#pragma unroll
    for (int nc = 0; nc < 4; ++nc) o[nc][r] *= inv;
  }
#pragma unroll
  for (int nc = 0; nc < 4; ++nc)
    store_acc_f16(aggr, HDIM, q0, hc + nc * 16, o[nc]);
}

// ---------------------------------------------------------------------------
// Kernel 4: output projection (aggr[N][512] @ wo[64][512]^T) + bo + residual
//           + LayerNorm2 -> h2 f16.  grid = N/16, one wave per block.
// ---------------------------------------------------------------------------
__global__ void k_proj_ln2(const _Float16* __restrict__ aggr,
                           const _Float16* __restrict__ wo,
                           const float* __restrict__ bo,
                           const float* __restrict__ x,
                           const float* __restrict__ g2,
                           const float* __restrict__ b2,
                           _Float16* __restrict__ h2) {
  const int lane = threadIdx.x & 31;
  const int n    = lane & 15;
  const int mo   = (lane >> 4) * 8;
  const int row0 = blockIdx.x * 16;

  v8f acc[4] = {{}, {}, {}, {}};
#pragma unroll 4
  for (int kc = 0; kc < 16; ++kc) {
    v16h a = frag_rowmajor(aggr, HDIM, row0, kc * 32);
#pragma unroll
    for (int nc = 0; nc < 4; ++nc) {
      v16h b = frag_rowmajor(wo, HDIM, nc * 16, kc * 32);  // B[k][j] = wo[j][k]
      acc[nc] = wmma_f16(a, b, acc[nc]);
    }
  }

  // h1 = proj + bo + x, then LN over the 64 columns of each row
  float h1[4][8];
#pragma unroll
  for (int nc = 0; nc < 4; ++nc) {
    float bj = bo[nc * 16 + n];
#pragma unroll
    for (int r = 0; r < 8; ++r)
      h1[nc][r] = acc[nc][r] + bj + x[(size_t)(row0 + mo + r) * DIM + nc * 16 + n];
  }
#pragma unroll
  for (int r = 0; r < 8; ++r) {
    float p = h1[0][r] + h1[1][r] + h1[2][r] + h1[3][r];
    float mean = hsum16(p) * (1.0f / 64.0f);
    float q2 = 0.0f;
#pragma unroll
    for (int nc = 0; nc < 4; ++nc) { float d = h1[nc][r] - mean; q2 += d * d; }
    float rs = rsqrtf(hsum16(q2) * (1.0f / 64.0f) + 1e-5f);
#pragma unroll
    for (int nc = 0; nc < 4; ++nc) {
      int j = nc * 16 + n;
      h2[(size_t)(row0 + mo + r) * DIM + j] =
          (_Float16)((h1[nc][r] - mean) * rs * g2[j] + b2[j]);
    }
  }
}

// ---------------------------------------------------------------------------
// Kernel 5: FFN.  t = relu(h2 @ W1^T + bf1); out = 2*(t @ W2^T + bf2).
// grid = N/16, one wave per block, ReLU tile staged through LDS.
// ---------------------------------------------------------------------------
__global__ void k_ffn(const _Float16* __restrict__ h2,
                      const _Float16* __restrict__ w1,
                      const _Float16* __restrict__ w2,
                      const float* __restrict__ bf1,
                      const float* __restrict__ bf2,
                      float* __restrict__ out) {
  __shared__ _Float16 lds[16 * 64];
  const int lane = threadIdx.x & 31;
  const int n    = lane & 15;
  const int mo   = (lane >> 4) * 8;
  const int row0 = blockIdx.x * 16;

  v8f t[4] = {{}, {}, {}, {}};
#pragma unroll
  for (int kc = 0; kc < 2; ++kc) {
    v16h a = frag_rowmajor(h2, DIM, row0, kc * 32);
#pragma unroll
    for (int nc = 0; nc < 4; ++nc)
      t[nc] = wmma_f16(a, frag_rowmajor(w1, DIM, nc * 16, kc * 32), t[nc]);
  }
#pragma unroll
  for (int nc = 0; nc < 4; ++nc) {
    float bj = bf1[nc * 16 + n];
#pragma unroll
    for (int r = 0; r < 8; ++r)
      lds[(mo + r) * 64 + nc * 16 + n] = (_Float16)fmaxf(t[nc][r] + bj, 0.0f);
  }
  __syncthreads();

  v8f f[4] = {{}, {}, {}, {}};
#pragma unroll
  for (int kc = 0; kc < 2; ++kc) {
    v16h a = frag_rowmajor(lds, DIM, 0, kc * 32);
#pragma unroll
    for (int nc = 0; nc < 4; ++nc)
      f[nc] = wmma_f16(a, frag_rowmajor(w2, DIM, nc * 16, kc * 32), f[nc]);
  }
#pragma unroll
  for (int nc = 0; nc < 4; ++nc) {
    float bj = bf2[nc * 16 + n];
#pragma unroll
    for (int r = 0; r < 8; ++r)
      out[(size_t)(row0 + mo + r) * DIM + nc * 16 + n] = 2.0f * (f[nc][r] + bj);
  }
}

// ---------------------------------------------------------------------------
// Host-side launch
// ---------------------------------------------------------------------------
extern "C" void kernel_launch(void* const* d_in, const int* in_sizes, int n_in,
                              void* d_out, int out_size, void* d_ws, size_t ws_size,
                              hipStream_t stream) {
  // setup_inputs() order:
  const float* x      = (const float*)d_in[0];
  const float* coords = (const float*)d_in[1];
  const float* sq     = (const float*)d_in[2];
  const float* sk     = (const float*)d_in[3];
  const float* sv     = (const float*)d_in[4];
  const float* s_rpe  = (const float*)d_in[5];
  const float* g1     = (const float*)d_in[6];
  const float* beta1  = (const float*)d_in[7];
  const float* g2     = (const float*)d_in[8];
  const float* beta2  = (const float*)d_in[9];
  const float* wo     = (const float*)d_in[10];
  const float* bo     = (const float*)d_in[11];
  const float* s1     = (const float*)d_in[12];
  const float* bf1    = (const float*)d_in[13];
  const float* s2     = (const float*)d_in[14];
  const float* bf2    = (const float*)d_in[15];
  const int* wq_int   = (const int*)d_in[16];
  const int* wk_int   = (const int*)d_in[17];
  const int* wv_int   = (const int*)d_in[18];
  const int* wrpe_int = (const int*)d_in[19];
  const int* w1_int   = (const int*)d_in[20];
  const int* w2_int   = (const int*)d_in[21];

  // workspace carve-up (f16 unless noted), 256B aligned
  char* ws = (char*)d_ws;
  size_t off = 0;
  auto carve = [&](size_t bytes) {
    char* p = ws + off;
    off = (off + bytes + 255) & ~(size_t)255;
    return p;
  };
  _Float16* wq_h   = (_Float16*)carve(HDIM * DIM * 2);
  _Float16* wk_h   = (_Float16*)carve(HDIM * DIM * 2);
  _Float16* wv_h   = (_Float16*)carve(HDIM * DIM * 2);
  _Float16* wo_h   = (_Float16*)carve(DIM * HDIM * 2);
  _Float16* w1_h   = (_Float16*)carve(DIM * DIM * 2);
  _Float16* w2_h   = (_Float16*)carve(DIM * DIM * 2);
  float*    wrpe_w = (float*)   carve(HEADS * 2 * 4);
  _Float16* xn_h   = (_Float16*)carve((size_t)NP * DIM * 2);
  _Float16* q_h    = (_Float16*)carve((size_t)NP * HDIM * 2);
  _Float16* k_h    = (_Float16*)carve((size_t)NP * HDIM * 2);
  _Float16* vt_h   = (_Float16*)carve((size_t)NP * HDIM * 2);  // [HD][N]
  _Float16* ag_h   = (_Float16*)carve((size_t)NP * HDIM * 2);
  _Float16* h2_h   = (_Float16*)carve((size_t)NP * DIM * 2);
  (void)ws_size; (void)n_in; (void)in_sizes; (void)out_size;

  // 0) dequant / convert weights (1/sqrt(64)=0.125 folded into Wq)
  k_dequant<<<(HDIM * DIM + 255) / 256, 256, 0, stream>>>(wq_int, sq, 0.125f, wq_h, HDIM * DIM);
  k_dequant<<<(HDIM * DIM + 255) / 256, 256, 0, stream>>>(wk_int, sk, 1.0f, wk_h, HDIM * DIM);
  k_dequant<<<(HDIM * DIM + 255) / 256, 256, 0, stream>>>(wv_int, sv, 1.0f, wv_h, HDIM * DIM);
  k_dequant<<<(DIM * DIM + 255) / 256, 256, 0, stream>>>(w1_int, s1, 1.0f, w1_h, DIM * DIM);
  k_dequant<<<(DIM * DIM + 255) / 256, 256, 0, stream>>>(w2_int, s2, 1.0f, w2_h, DIM * DIM);
  k_f32_to_f16<<<(DIM * HDIM + 255) / 256, 256, 0, stream>>>(wo, wo_h, DIM * HDIM);
  k_rpe<<<1, 32, 0, stream>>>(wrpe_int, s_rpe, wrpe_w);

  // 1) LN1
  k_layernorm<<<NP / 8, 256, 0, stream>>>(x, g1, beta1, xn_h);

  // 2) Q/K/V projections (V stored transposed)
  k_qkv_gemm<<<dim3(NP / 16, HDIM / 64, 3), 32, 0, stream>>>(xn_h, wq_h, wk_h, wv_h,
                                                             q_h, k_h, vt_h);

  // 3) flash attention with RPE bias
  k_attention<<<dim3(NP / 16, HEADS), 32, 0, stream>>>(q_h, k_h, vt_h, coords,
                                                       wrpe_w, ag_h);

  // 4) output projection + residual + LN2
  k_proj_ln2<<<NP / 16, 32, 0, stream>>>(ag_h, wo_h, bo, x, g2, beta2, h2_h);

  // 5) FFN, out = 2*ff
  k_ffn<<<NP / 16, 32, 0, stream>>>(h2_h, w1_h, w2_h, bf1, bf2, (float*)d_out);
}
